// EgyptianLinearAttention_70635032150389
// MI455X (gfx1250) — matrix operand order repaired and checked by reading, compiled, not run on hardware
//
#include <hip/hip_runtime.h>

#define S_LEN 2048
#define BATCH 4
#define NHEADS 12
#define HD 64
#define DMODEL 768
#define BH (BATCH * NHEADS)
#define NSTRIDE 683  // number of s with s % 3 == 0 in [0, 2048)

typedef __attribute__((ext_vector_type(16))) __bf16 v16bf;
typedef __attribute__((ext_vector_type(8))) float v8f;

struct __align__(16) U4 { unsigned int x, y, z, w; };
struct __align__(8)  U2 { unsigned int x, y; };
struct __align__(16) F4 { float x, y, z, w; };

__device__ __forceinline__ unsigned short f2bf(float f) {
  union { float f; unsigned u; } c; c.f = f;
  unsigned r = c.u + 0x7FFFu + ((c.u >> 16) & 1u);  // round-to-nearest-even
  return (unsigned short)(r >> 16);
}
__device__ __forceinline__ float bf2f(unsigned short h) {
  union { unsigned u; float f; } c; c.u = (unsigned)h << 16;
  return c.f;
}
__device__ __forceinline__ float fmap_f(float x) {  // elu(x)+1
  return x > 0.0f ? x + 1.0f : __expf(x);
}
// Two 16B LDS loads -> one 16x(bf16) WMMA operand fragment
__device__ __forceinline__ v16bf ld_frag(const unsigned short* p0, const unsigned short* p1) {
  union { v16bf v; U4 q[2]; } u;
  u.q[0] = *(const U4*)p0;
  u.q[1] = *(const U4*)p1;
  return u.v;
}
// Bulk fp32->bf16: one v_perm_b32 packs the top halves of two floats
// (truncation; bf16 inputs feed an f32-accumulating WMMA, <=1ulp bf16 error).
__device__ __forceinline__ U2 pack4(const F4& f) {
  U2 r;
  r.x = __builtin_amdgcn_perm(__float_as_uint(f.y), __float_as_uint(f.x), 0x07060302u);
  r.y = __builtin_amdgcn_perm(__float_as_uint(f.w), __float_as_uint(f.z), 0x07060302u);
  return r;
}

// ---------------------------------------------------------------------------
// GEMM: C[M,N] = A[M,K] * W[N,K]^T   (fp32 in, bf16 WMMA, fp32 out)
// mode 0: plain fp32 store to C
// mode 1: fused QKV epilogue -> qf/kf/vv bf16 in [b,h,s,d] layout (fmap on q,k)
// Block tile 128x64, BK=32, 256 threads (8 waves), wave tile 32x32 (2x2 WMMA).
// Software-pipelined: register prefetch + double-buffered LDS, 1 barrier/chunk.
// ---------------------------------------------------------------------------
#define BM 128
#define BN 64
#define BK 32
#define LP 40  // LDS row stride in halves (80B, keeps 16B alignment, spreads banks)

__global__ __launch_bounds__(256) void gemm_bf16_nt(
    const float* __restrict__ A, const float* __restrict__ W,
    float* __restrict__ C, int M, int N, int K, int mode,
    unsigned short* __restrict__ qf, unsigned short* __restrict__ kf,
    unsigned short* __restrict__ vv) {
  __shared__ unsigned short sA[2][BM][LP];
  __shared__ unsigned short sB[2][BN][LP];
  const int tid  = threadIdx.x;
  const int lane = tid & 31;
  const int wave = tid >> 5;
  const int wm = wave & 3;   // 4 waves along M
  const int wn = wave >> 2;  // 2 waves along N
  const int mblocks = M / BM;
  const int bm = blockIdx.x % mblocks;
  const int bn = blockIdx.x / mblocks;
  const size_t row0 = (size_t)bm * BM;
  const size_t col0 = (size_t)bn * BN;
  const int lr  = tid >> 3;        // 0..31 loader row
  const int lc  = (tid & 7) * 4;   // loader col (floats)
  const int l16 = lane & 15;
  const int hi  = lane >> 4;
  const size_t Ks = (size_t)K;

  // per-thread load base pointers; row term folds into the 24-bit inst offset
  const float* ap = A + (row0 + lr) * Ks + lc;
  const float* wp = W + (col0 + lr) * Ks + lc;

  v8f acc[2][2] = {};
  F4 fa[4], fb[2];

  // prefetch chunk 0 (6 b128 loads in flight before any wait)
#pragma unroll
  for (int rr = 0; rr < 4; ++rr) fa[rr] = *(const F4*)(ap + (size_t)rr * 32 * Ks);
#pragma unroll
  for (int rr = 0; rr < 2; ++rr) fb[rr] = *(const F4*)(wp + (size_t)rr * 32 * Ks);

  const int nch = K / BK;
  int p = 0;
  for (int c = 0; c < nch; ++c) {
    // convert + store current chunk into LDS buffer p (2 v_perm per float4)
#pragma unroll
    for (int rr = 0; rr < 4; ++rr) *(U2*)&sA[p][lr + rr * 32][lc] = pack4(fa[rr]);
#pragma unroll
    for (int rr = 0; rr < 2; ++rr) *(U2*)&sB[p][lr + rr * 32][lc] = pack4(fb[rr]);
    // prefetch next chunk into registers (overlaps with WMMA below)
    if (c + 1 < nch) {
      ap += BK; wp += BK;
#pragma unroll
      for (int rr = 0; rr < 4; ++rr) fa[rr] = *(const F4*)(ap + (size_t)rr * 32 * Ks);
#pragma unroll
      for (int rr = 0; rr < 2; ++rr) fb[rr] = *(const F4*)(wp + (size_t)rr * 32 * Ks);
    }
    __syncthreads();
#pragma unroll
    for (int j = 0; j < 2; ++j) {
      const unsigned short* br = &sB[p][wn * 32 + j * 16 + l16][0];
      v16bf bf = ld_frag(br + hi * 8, br + hi * 8 + 16);
#pragma unroll
      for (int i = 0; i < 2; ++i) {
        const unsigned short* ar = &sA[p][wm * 32 + i * 16 + l16][0];
        v16bf af = ld_frag(ar + hi * 8, ar + hi * 8 + 16);
        acc[i][j] = __builtin_amdgcn_wmma_f32_16x16x32_bf16(
            false, af, false, bf, (short)0, acc[i][j], false, false);
      }
    }
    p ^= 1;
  }

  if (mode == 0) {
#pragma unroll
    for (int i = 0; i < 2; ++i)
#pragma unroll
      for (int j = 0; j < 2; ++j) {
        size_t r0 = row0 + wm * 32 + i * 16 + hi * 8;
        size_t c0 = col0 + wn * 32 + j * 16 + l16;
#pragma unroll
        for (int t = 0; t < 8; ++t) C[(r0 + t) * N + c0] = acc[i][j][t];
      }
  } else {
    // fused QKV epilogue: col = which*768 + h*64 + d ; row = b*2048 + s
#pragma unroll
    for (int i = 0; i < 2; ++i)
#pragma unroll
      for (int j = 0; j < 2; ++j) {
        size_t rbase = row0 + wm * 32 + i * 16 + hi * 8;
        int cg = (int)col0 + wn * 32 + j * 16 + l16;
        int which = cg / DMODEL;
        int rem = cg % DMODEL;
        int hh = rem >> 6;
        int d = rem & 63;
        unsigned short* dst = (which == 0) ? qf : ((which == 1) ? kf : vv);
#pragma unroll
        for (int t = 0; t < 8; ++t) {
          size_t r = rbase + t;
          int b = (int)(r >> 11);       // /2048
          int s = (int)(r & 2047);
          float val = acc[i][j][t];
          if (which != 2) val = fmap_f(val);
          dst[(((size_t)(b * NHEADS + hh)) * S_LEN + s) * HD + d] = f2bf(val);
        }
      }
  }
}

// ---------------------------------------------------------------------------
// Per-(b,h) sums of k_f over s (full and stride-3 sampled)
// ---------------------------------------------------------------------------
__global__ __launch_bounds__(64) void ksum_kernel(
    const unsigned short* __restrict__ kf, float* __restrict__ ksg,
    float* __restrict__ kss) {
  int bh = blockIdx.x;
  int d = threadIdx.x;
  const unsigned short* p = kf + (size_t)bh * S_LEN * HD + d;
  float sg = 0.f, ss = 0.f;
  for (int s = 0; s < S_LEN; ++s) {
    float x = bf2f(p[(size_t)s * HD]);
    sg += x;
    if (s % 3 == 0) ss += x;
  }
  ksg[bh * HD + d] = sg;
  kss[bh * HD + d] = ss;
}

// ---------------------------------------------------------------------------
// KV[64,64] = k_f^T @ v per (b,h) via WMMA. strided=1 -> s = 3j, j<683 (zero pad)
// 128 threads = 4 waves; wave w owns 16 columns of KV; 4 accumulators each.
// Transpose happens while staging 32x64 chunks into LDS (ping-pong buffered).
// ---------------------------------------------------------------------------
struct Chunk16 { U4 q[2]; };

__device__ __forceinline__ void load_chunk(const unsigned short* base, size_t srow,
                                           int dg, bool ok, Chunk16& out) {
  if (ok) {
    const unsigned short* p = base + srow * HD + dg * 16;
    out.q[0] = *(const U4*)p;
    out.q[1] = *(const U4*)(p + 8);
  } else {
    U4 z = {0u, 0u, 0u, 0u};
    out.q[0] = z; out.q[1] = z;
  }
}

__global__ __launch_bounds__(128) void kv_wmma(
    const unsigned short* __restrict__ kf, const unsigned short* __restrict__ vv,
    float* __restrict__ KV, int strided) {
  __shared__ unsigned short kT[2][64][LP];
  __shared__ unsigned short vT[2][64][LP];
  const int bh = blockIdx.x;
  const int tid = threadIdx.x;
  const int lane = tid & 31;
  const int wave = tid >> 5;  // e-tile
  const int l16 = lane & 15;
  const int hi = lane >> 4;
  const int sl = tid & 31;    // s_local in chunk
  const int dg = tid >> 5;    // d-group (16 halves)
  const unsigned short* kb = kf + (size_t)bh * S_LEN * HD;
  const unsigned short* vb = vv + (size_t)bh * S_LEN * HD;
  const int nch = strided ? 22 : (S_LEN / 32);

  v8f acc[4] = {};
  Chunk16 ka, va, ka2, va2;

  {
    int sidx = sl;
    size_t srow = strided ? (size_t)(3 * sidx) : (size_t)sidx;
    bool ok = strided ? (sidx < NSTRIDE) : true;
    load_chunk(kb, srow, dg, ok, ka);
    load_chunk(vb, srow, dg, ok, va);
  }

  int p = 0;
  for (int c = 0; c < nch; ++c) {
    // transposed store of current chunk into buffer p
    union { Chunk16 c; unsigned short s[16]; } uk, uv;
    uk.c = ka; uv.c = va;
#pragma unroll
    for (int t = 0; t < 16; ++t) {
      kT[p][dg * 16 + t][sl] = uk.s[t];
      vT[p][dg * 16 + t][sl] = uv.s[t];
    }
    if (c + 1 < nch) {
      int sidx = (c + 1) * 32 + sl;
      size_t srow = strided ? (size_t)(3 * sidx) : (size_t)sidx;
      bool ok = strided ? (sidx < NSTRIDE) : true;
      load_chunk(kb, srow, dg, ok, ka2);
      load_chunk(vb, srow, dg, ok, va2);
    }
    __syncthreads();
    const unsigned short* br = &vT[p][wave * 16 + l16][0];
    v16bf bf = ld_frag(br + hi * 8, br + hi * 8 + 16);
#pragma unroll
    for (int i = 0; i < 4; ++i) {
      const unsigned short* ar = &kT[p][i * 16 + l16][0];
      v16bf af = ld_frag(ar + hi * 8, ar + hi * 8 + 16);
      acc[i] = __builtin_amdgcn_wmma_f32_16x16x32_bf16(
          false, af, false, bf, (short)0, acc[i], false, false);
    }
    ka = ka2; va = va2;
    p ^= 1;
  }
  float* o = KV + (size_t)bh * 64 * 64;
#pragma unroll
  for (int i = 0; i < 4; ++i) {
    int r0 = i * 16 + hi * 8;
    int c0 = wave * 16 + l16;
#pragma unroll
    for (int t = 0; t < 8; ++t) o[(r0 + t) * 64 + c0] = acc[i][t];
  }
}

// ---------------------------------------------------------------------------
// Anchor band head: anchor_out[b,h,a,e] = (fmap(aq) @ KVg) / clip(fmap(aq)·ksum)
// Also writes aq_f (fmapped anchors, per head) once for reuse in band_combine.
// ---------------------------------------------------------------------------
__global__ __launch_bounds__(64) void anchor_kernel(
    const float* __restrict__ anchor_q, const float* __restrict__ KVg,
    const float* __restrict__ ksg, float* __restrict__ anchor_out,
    float* __restrict__ aqf) {
  const int bh = blockIdx.x;
  const int h = bh % NHEADS;
  const int b = bh / NHEADS;
  const int e = threadIdx.x;
  __shared__ float saq[12][64];
  for (int a = 0; a < 12; ++a)
    saq[a][e] = fmap_f(anchor_q[(size_t)h * DMODEL + a * HD + e]);
  __syncthreads();
  if (b == 0)
    for (int a = 0; a < 12; ++a)
      aqf[((size_t)h * 12 + a) * HD + e] = saq[a][e];
  const float* kv = KVg + (size_t)bh * 4096;
  const float* ks = ksg + (size_t)bh * HD;
  for (int a = 0; a < 12; ++a) {
    float accv = 0.f, nrm = 0.f;
    for (int d2 = 0; d2 < 64; ++d2) {
      accv += saq[a][d2] * kv[d2 * 64 + e];
      nrm  += saq[a][d2] * ks[d2];
    }
    anchor_out[(size_t)bh * 768 + a * HD + e] = accv / fmaxf(nrm, 1e-6f);
  }
}

// ---------------------------------------------------------------------------
// Fused band combine: local window + stride matvec + global anchors,
// softmax-weighted, written as attn[b,s,h*64+e] (ready for out projection).
// Block = one (b,h) x 4 consecutive s; 256 threads = 4 s x 64 e.
// Out-of-range local keys behave as fmap(0)=1 (ones-dot), values as zero,
// matching the reference's pad-then-fmap semantics.
// ---------------------------------------------------------------------------
__global__ __launch_bounds__(256) void band_combine(
    const unsigned short* __restrict__ qf, const unsigned short* __restrict__ kf,
    const unsigned short* __restrict__ vv,
    const float* __restrict__ KVs, const float* __restrict__ kss,
    const float* __restrict__ anchor_out, const float* __restrict__ aqf,
    const float* __restrict__ wlp, const float* __restrict__ wsp,
    const float* __restrict__ wgp, float* __restrict__ attn) {
  const int nsc = S_LEN / 4;
  const int sc = blockIdx.x % nsc;
  const int bh = blockIdx.x / nsc;
  const int h = bh % NHEADS;
  const int b = bh / NHEADS;
  const int s0 = sc * 4;
  const int tid = threadIdx.x;
  const int sl = tid >> 6;
  const int e = tid & 63;

  __shared__ float sq[4][64];
  __shared__ float sKV[64][64];
  __shared__ float sks[64];
  __shared__ float sao[12][64];
  __shared__ float qk_loc[4][12];
  __shared__ float qk_glb[4][12];

  sq[sl][e] = bf2f(qf[((size_t)bh * S_LEN + s0 + sl) * HD + e]);
  const float* kvp = KVs + (size_t)bh * 4096;
  for (int i = tid; i < 4096; i += 256) sKV[i >> 6][i & 63] = kvp[i];
  if (tid < 64) sks[tid] = kss[(size_t)bh * HD + tid];
  for (int i = tid; i < 768; i += 256) sao[i / 64][i & 63] = anchor_out[(size_t)bh * 768 + i];
  __syncthreads();

  if (tid < 48) {  // local-window qk dots (4 s x 12 w)
    int ss = tid / 12, w = tid % 12;
    int sk = s0 + ss + w - 6;
    float dot = 0.f;
    if (sk >= 0 && sk < S_LEN) {
      const unsigned short* kp = kf + ((size_t)bh * S_LEN + sk) * HD;
      for (int d2 = 0; d2 < 64; ++d2) dot += sq[ss][d2] * bf2f(kp[d2]);
    } else {
      for (int d2 = 0; d2 < 64; ++d2) dot += sq[ss][d2];  // key = ones
    }
    qk_loc[ss][w] = dot;
  } else if (tid < 96) {  // anchor qk dots (4 s x 12 a)
    int t2 = tid - 48;
    int ss = t2 / 12, a = t2 % 12;
    const float* ap = aqf + ((size_t)h * 12 + a) * HD;
    float dot = 0.f;
    for (int d2 = 0; d2 < 64; ++d2) dot += sq[ss][d2] * ap[d2];
    qk_glb[ss][a] = dot;
  }
  __syncthreads();

  // softmax of the three band weights
  float a0 = *wlp, a1 = *wsp, a2 = *wgp;
  float mx = fmaxf(a0, fmaxf(a1, a2));
  float e0 = __expf(a0 - mx), e1 = __expf(a1 - mx), e2 = __expf(a2 - mx);
  float inv = 1.f / (e0 + e1 + e2);

  // local band
  float out_l = 0.f, nl = 0.f;
#pragma unroll
  for (int w = 0; w < 12; ++w) {
    float qk = qk_loc[sl][w];
    nl += qk;
    int sk = s0 + sl + w - 6;
    if (sk >= 0 && sk < S_LEN)
      out_l += qk * bf2f(vv[((size_t)bh * S_LEN + sk) * HD + e]);
  }
  out_l /= fmaxf(nl, 1e-6f);

  // stride band: q_f . KV_s column e
  float out_s = 0.f, ns = 0.f;
  for (int d2 = 0; d2 < 64; ++d2) {
    float qd = sq[sl][d2];
    out_s += qd * sKV[d2][e];
    ns += qd * sks[d2];
  }
  out_s /= fmaxf(ns, 1e-6f);

  // global band
  float out_g = 0.f, ng = 0.f;
#pragma unroll
  for (int a = 0; a < 12; ++a) {
    float qk = qk_glb[sl][a];
    ng += qk;
    out_g += qk * sao[a][e];
  }
  out_g /= fmaxf(ng, 1e-6f);

  float res = (e0 * out_l + e1 * out_s + e2 * out_g) * inv;
  attn[((size_t)(b * S_LEN + s0 + sl)) * DMODEL + h * HD + e] = res;
}

// ---------------------------------------------------------------------------
extern "C" void kernel_launch(void* const* d_in, const int* in_sizes, int n_in,
                              void* d_out, int out_size, void* d_ws, size_t ws_size,
                              hipStream_t stream) {
  (void)in_sizes; (void)n_in; (void)out_size; (void)ws_size;
  const float* x        = (const float*)d_in[0];  // [4,2048,768]
  const float* qkv_w    = (const float*)d_in[1];  // [2304,768]
  const float* out_w    = (const float*)d_in[2];  // [768,768]
  const float* anchor_q = (const float*)d_in[3];  // [1,12,768]
  const float* wlp      = (const float*)d_in[4];
  const float* wsp      = (const float*)d_in[5];
  const float* wgp      = (const float*)d_in[6];
  float* outp = (float*)d_out;

  char* ws = (char*)d_ws;
  const size_t BF = (size_t)BH * S_LEN * HD * sizeof(unsigned short);  // 12.58 MB
  size_t off = 0;
  unsigned short* qf = (unsigned short*)(ws + off); off += BF;
  unsigned short* kf = (unsigned short*)(ws + off); off += BF;
  unsigned short* vv = (unsigned short*)(ws + off); off += BF;
  float* attn = (float*)(ws + off); off += (size_t)BATCH * S_LEN * DMODEL * sizeof(float);
  float* KVg  = (float*)(ws + off); off += (size_t)BH * 4096 * sizeof(float);
  float* KVs  = (float*)(ws + off); off += (size_t)BH * 4096 * sizeof(float);
  float* ksg  = (float*)(ws + off); off += (size_t)BH * HD * sizeof(float);
  float* kss  = (float*)(ws + off); off += (size_t)BH * HD * sizeof(float);
  float* aout = (float*)(ws + off); off += (size_t)BH * 768 * sizeof(float);
  float* aqf  = (float*)(ws + off); off += (size_t)NHEADS * 12 * HD * sizeof(float);

  const int M = BATCH * S_LEN;  // 8192

  // 1) QKV projection (bf16 WMMA), fused fmap + [b,h,s,d] bf16 epilogue
  gemm_bf16_nt<<<dim3((M / BM) * ((3 * DMODEL) / BN)), 256, 0, stream>>>(
      x, qkv_w, attn /*unused*/, M, 3 * DMODEL, DMODEL, 1, qf, kf, vv);

  // 2) k_f sums (full + strided)
  ksum_kernel<<<dim3(BH), 64, 0, stream>>>(kf, ksg, kss);

  // 3) KV matrices via WMMA (global band + stride band)
  kv_wmma<<<dim3(BH), 128, 0, stream>>>(kf, vv, KVg, 0);
  kv_wmma<<<dim3(BH), 128, 0, stream>>>(kf, vv, KVs, 1);

  // 4) anchor outputs
  anchor_kernel<<<dim3(BH), 64, 0, stream>>>(anchor_q, KVg, ksg, aout, aqf);

  // 5) fused three-band combine -> attn [B,S,D]
  band_combine<<<dim3(BH * (S_LEN / 4)), 256, 0, stream>>>(
      qf, kf, vv, KVs, kss, aout, aqf, wlp, wsp, wgp, attn);

  // 6) output projection (bf16 WMMA) -> d_out
  gemm_bf16_nt<<<dim3((M / BM) * (DMODEL / BN)), 256, 0, stream>>>(
      attn, out_w, outp, M, DMODEL, DMODEL, 0, qf, kf, vv);
}
